// MultiLatentAttention_88081189307138
// MI455X (gfx1250) — compile-verified, matrix operand order checked by reading, not run on hardware
//
#include <hip/hip_runtime.h>
#include <stdint.h>

// ---------------- problem constants ----------------
#define B_  2
#define S_  2048
#define D_  2048
#define H_  16
#define DK_ 128
#define DV_ 128
#define L_  64

typedef __attribute__((ext_vector_type(16))) __bf16 v16bf;
typedef __attribute__((ext_vector_type(8)))  __bf16 bf16x8;
typedef __attribute__((ext_vector_type(4)))  __bf16 bf16x4;
typedef __attribute__((ext_vector_type(8)))  float  v8f;
typedef __attribute__((ext_vector_type(4)))  float  f32x4;
typedef __attribute__((ext_vector_type(4)))  int    v4i;

static __device__ __forceinline__ v16bf combine8(bf16x8 lo, bf16x8 hi) {
  return __builtin_shufflevector(lo, hi, 0,1,2,3,4,5,6,7,8,9,10,11,12,13,14,15);
}
static __device__ __forceinline__ v8f wmma_bf16(v16bf a, v16bf b, v8f c) {
  // D = A(16x32) * B(32x16) + C, f32 accumulate
  return __builtin_amdgcn_wmma_f32_16x16x32_bf16(false, a, false, b, (short)0, c, false, false);
}

// -------- CDNA5 async global->LDS copy (ASYNCcnt-tracked, no VGPR data) ----
static __device__ __forceinline__ void async_copy_b128(void* lds_dst, const void* gsrc) {
#if __has_builtin(__builtin_amdgcn_global_load_async_to_lds_b128)
  __builtin_amdgcn_global_load_async_to_lds_b128(
      (__attribute__((address_space(1))) v4i*)(uintptr_t)gsrc,
      (__attribute__((address_space(3))) v4i*)(unsigned)(uintptr_t)lds_dst,
      0, 0);
#else
  asm volatile("global_load_async_to_lds_b128 %0, %1, off"
               :: "v"((unsigned)(uintptr_t)lds_dst), "v"(gsrc)
               : "memory");
#endif
}
static __device__ __forceinline__ void wait_async0() {
#if __has_builtin(__builtin_amdgcn_s_wait_asynccnt)
  __builtin_amdgcn_s_wait_asynccnt(0);
#else
  asm volatile("s_wait_asynccnt 0" ::: "memory");
#endif
}

// ---------------- fp32 -> bf16 conversion (activations) ----------------
__global__ __launch_bounds__(256)
void cvt_bf16_kernel(const float* __restrict__ x, __bf16* __restrict__ y, int n4) {
  int i = blockIdx.x * blockDim.x + threadIdx.x;
  if (i >= n4) return;
  f32x4 v = ((const f32x4*)x)[i];
  bf16x4 o;
  o.x = (__bf16)v.x; o.y = (__bf16)v.y; o.z = (__bf16)v.z; o.w = (__bf16)v.w;
  ((bf16x4*)y)[i] = o;
}

// ------- fp32 [K,N] -> bf16 transposed [N,K] (one-time weight prep) -------
__global__ __launch_bounds__(256)
void cvt_transpose_kernel(const float* __restrict__ W, __bf16* __restrict__ WT,
                          int K, int N) {
  __shared__ float tile[32][33];
  const int tx = threadIdx.x & 31;
  const int ty = threadIdx.x >> 5;             // 0..7
  const int kb = blockIdx.y * 32, nb = blockIdx.x * 32;
  #pragma unroll
  for (int i = 0; i < 4; i++)
    tile[ty + i * 8][tx] = W[(long)(kb + ty + i * 8) * N + nb + tx];
  __syncthreads();
  #pragma unroll
  for (int i = 0; i < 4; i++)
    WT[(long)(nb + ty + i * 8) * K + kb + tx] = (__bf16)tile[tx][ty + i * 8];
}

// ---------------- generic bf16 WMMA GEMM (async double-buffered) ----------
// C[M,N] = A[M,K] @ W[K,N] + bias, with W supplied pre-transposed WT[N,K].
// Batched via blockIdx.z: aOff=(z/hDiv)*aSB+(z%hDiv)*aSH ; cOff=z*cSZ.
// tStore: store element (row,col) at cOff + col*ldc + row (materialize C^T).
// Cf!=null -> f32 output, else bf16 to Cb.
__global__ __launch_bounds__(256)
void gemm_bf16_kernel(const __bf16* __restrict__ A, const __bf16* __restrict__ WT,
                      const float* __restrict__ bias,
                      __bf16* __restrict__ Cb, float* __restrict__ Cf,
                      int M, int N, int K, int lda, int ldc,
                      long aSB, long aSH, int hDiv, long cSZ, int tStore)
{
  __shared__ __bf16 lgA[2][128 * 32];      // [buf][row][k]  2 x 8 KB
  __shared__ __bf16 lgB[2][128 * 32];      // [buf][n][k]    2 x 8 KB
  const int tid  = threadIdx.x;
  const int lane = tid & 31, wid = tid >> 5;
  const int hl   = lane >> 4, ln = lane & 15;
  const int wm   = wid & 3,  wn  = wid >> 2;        // 4 waves on M, 2 on N
  const int z    = blockIdx.z;
  const long aOff = (long)(z / hDiv) * aSB + (long)(z % hDiv) * aSH;
  const long cOff = (long)z * cSZ;
  const int mBase = blockIdx.y * 128;
  const int nBase = blockIdx.x * 128;
  const long aBase = aOff + (long)mBase * lda;
  const long bBase = (long)nBase * K;               // WT row stride = K

  const v8f zero8 = {0.f,0.f,0.f,0.f,0.f,0.f,0.f,0.f};
  v8f acc[2][4];
  #pragma unroll
  for (int i = 0; i < 2; i++)
    #pragma unroll
    for (int j = 0; j < 4; j++) acc[i][j] = zero8;

  // async-stage one 128x32 A tile and 128x32 B tile into buffer `buf`
  auto stage = [&](int buf, int k0) {
    #pragma unroll
    for (int i = 0; i < 2; i++) {
      int chunk = tid + i * 256;                    // 512 chunks of 8 bf16
      int row = chunk >> 2, kc8 = (chunk & 3) * 8;
      async_copy_b128(&lgA[buf][row * 32 + kc8],
                      A + aBase + (long)row * lda + k0 + kc8);
      async_copy_b128(&lgB[buf][row * 32 + kc8],
                      WT + bBase + (long)row * K + k0 + kc8);
    }
  };

  stage(0, 0);
  wait_async0();
  __syncthreads();

  for (int k0 = 0; k0 < K; k0 += 32) {
    const int cur = (k0 >> 5) & 1;
    if (k0 + 32 < K) stage(cur ^ 1, k0 + 32);       // async prefetch of next tile

    v16bf af[2], bfrag[4];
    #pragma unroll
    for (int mf = 0; mf < 2; mf++) {
      int r = wm * 32 + mf * 16 + ln;
      bf16x8 lo = *(const bf16x8*)&lgA[cur][r * 32 + 8 * hl];
      bf16x8 hi = *(const bf16x8*)&lgA[cur][r * 32 + 16 + 8 * hl];
      af[mf] = combine8(lo, hi);
    }
    #pragma unroll
    for (int nf = 0; nf < 4; nf++) {
      int c = wn * 64 + nf * 16 + ln;
      bf16x8 lo = *(const bf16x8*)&lgB[cur][c * 32 + 8 * hl];
      bf16x8 hi = *(const bf16x8*)&lgB[cur][c * 32 + 16 + 8 * hl];
      bfrag[nf] = combine8(lo, hi);
    }
    #pragma unroll
    for (int mf = 0; mf < 2; mf++)
      #pragma unroll
      for (int nf = 0; nf < 4; nf++)
        acc[mf][nf] = wmma_bf16(af[mf], bfrag[nf], acc[mf][nf]);

    wait_async0();        // next-tile async copies landed
    __syncthreads();      // all waves done reading `cur` before it is re-staged
  }

  // epilogue: bias + store (D layout: half -> rows 8h..8h+7, col = lane&15)
  #pragma unroll
  for (int mf = 0; mf < 2; mf++)
    #pragma unroll
    for (int nf = 0; nf < 4; nf++) {
      int col = nBase + wn * 64 + nf * 16 + ln;
      float bv = bias ? bias[col] : 0.0f;
      #pragma unroll
      for (int r = 0; r < 8; r++) {
        int row = mBase + wm * 32 + mf * 16 + 8 * hl + r;
        float val = acc[mf][nf][r] + bv;
        if (Cf)          Cf[cOff + (long)row * ldc + col] = val;
        else if (tStore) Cb[cOff + (long)col * ldc + row] = (__bf16)val;
        else             Cb[cOff + (long)row * ldc + col] = (__bf16)val;
      }
    }
}

// ---------------- causal flash attention (bf16 WMMA, f32 softmax) --------
// Q:  [B*S, H*DK] bf16 ; Kt: [B,H,S,DK] bf16 ; Vt: [B,H,DV,S] bf16 (V^T)
// O:  [B*S, H*DV] bf16
__global__ __launch_bounds__(128)
void mla_flash_kernel(const __bf16* __restrict__ Q, const __bf16* __restrict__ Kt,
                      const __bf16* __restrict__ Vt, __bf16* __restrict__ O)
{
  __shared__ __bf16 Pbuf[4][16 * 32];     // per-wave P staging (D->A layout transpose)
  const int tid  = threadIdx.x;
  const int lane = tid & 31, wid = tid >> 5;
  const int hl   = lane >> 4, ln = lane & 15;
  const int z = blockIdx.y;               // b*H + h
  const int b = z >> 4, h = z & 15;
  const int qbase = blockIdx.x * 64 + wid * 16;   // 16 query rows per wave
  const __bf16* Kb = Kt + (long)z * S_ * DK_;
  const __bf16* Vb = Vt + (long)z * DV_ * S_;

  const v8f zero8 = {0.f,0.f,0.f,0.f,0.f,0.f,0.f,0.f};

  v16bf qf[4];
  {
    const __bf16* qrow = Q + (long)(b * S_ + qbase + ln) * (H_ * DK_) + h * DK_;
    #pragma unroll
    for (int kc = 0; kc < 4; kc++) {
      bf16x8 lo = *(const bf16x8*)(qrow + kc * 32 + 8 * hl);
      bf16x8 hi = *(const bf16x8*)(qrow + kc * 32 + 16 + 8 * hl);
      qf[kc] = combine8(lo, hi);
    }
  }

  v8f o[8];
  float mi[8], li[8];
  #pragma unroll
  for (int t = 0; t < 8; t++) o[t] = zero8;
  #pragma unroll
  for (int r = 0; r < 8; r++) { mi[r] = -__builtin_inff(); li[r] = 0.0f; }

  const float scale = 0.08838834764831845f;  // 1/sqrt(128)

  for (int kb = 0; kb <= qbase + 15; kb += 32) {
    if (kb + 32 <= qbase + 15)
      __builtin_prefetch(Kb + (long)(kb + 32 + ln) * DK_, 0, 1);

    v8f s[2]; s[0] = zero8; s[1] = zero8;
    #pragma unroll
    for (int t = 0; t < 2; t++) {
      const __bf16* krow = Kb + (long)(kb + t * 16 + ln) * DK_;
      #pragma unroll
      for (int kc = 0; kc < 4; kc++) {
        bf16x8 lo = *(const bf16x8*)(krow + kc * 32 + 8 * hl);
        bf16x8 hi = *(const bf16x8*)(krow + kc * 32 + 16 + 8 * hl);
        s[t] = wmma_bf16(qf[kc], combine8(lo, hi), s[t]);
      }
    }

    float ps[2][8];
    #pragma unroll
    for (int t = 0; t < 2; t++)
      #pragma unroll
      for (int r = 0; r < 8; r++) {
        int col = kb + t * 16 + ln;
        int row = qbase + 8 * hl + r;
        float v = s[t][r] * scale;
        ps[t][r] = (col > row) ? -__builtin_inff() : v;
      }

    #pragma unroll
    for (int r = 0; r < 8; r++) {
      float rm = fmaxf(ps[0][r], ps[1][r]);
      #pragma unroll
      for (int off = 1; off < 16; off <<= 1) rm = fmaxf(rm, __shfl_xor(rm, off, 32));
      float mnew  = fmaxf(mi[r], rm);
      float alpha = __expf(mi[r] - mnew);
      float p0 = __expf(ps[0][r] - mnew);
      float p1 = __expf(ps[1][r] - mnew);
      float rs = p0 + p1;
      #pragma unroll
      for (int off = 1; off < 16; off <<= 1) rs += __shfl_xor(rs, off, 32);
      li[r] = li[r] * alpha + rs;
      mi[r] = mnew;
      ps[0][r] = p0; ps[1][r] = p1;
      #pragma unroll
      for (int t = 0; t < 8; t++) o[t][r] *= alpha;
    }

    __bf16* P = &Pbuf[wid][0];
    #pragma unroll
    for (int t = 0; t < 2; t++)
      #pragma unroll
      for (int r = 0; r < 8; r++)
        P[(8 * hl + r) * 32 + t * 16 + ln] = (__bf16)ps[t][r];
    asm volatile("s_wait_dscnt 0" ::: "memory");   // CDNA5 split DS counter wait
    v16bf pf;
    {
      bf16x8 lo = *(const bf16x8*)&P[ln * 32 + 8 * hl];
      bf16x8 hi = *(const bf16x8*)&P[ln * 32 + 16 + 8 * hl];
      pf = combine8(lo, hi);
    }

    #pragma unroll
    for (int t = 0; t < 8; t++) {
      const __bf16* vrow = Vb + (long)(t * 16 + ln) * S_ + kb;
      bf16x8 lo = *(const bf16x8*)(vrow + 8 * hl);
      bf16x8 hi = *(const bf16x8*)(vrow + 16 + 8 * hl);
      o[t] = wmma_bf16(pf, combine8(lo, hi), o[t]);
    }
  }

  #pragma unroll
  for (int t = 0; t < 8; t++)
    #pragma unroll
    for (int r = 0; r < 8; r++) {
      int row = qbase + 8 * hl + r;
      float val = o[t][r] / li[r];
      O[(long)(b * S_ + row) * (H_ * DV_) + h * DV_ + t * 16 + ln] = (__bf16)val;
    }
}

// ---------------- host orchestration ----------------
extern "C" void kernel_launch(void* const* d_in, const int* in_sizes, int n_in,
                              void* d_out, int out_size, void* d_ws, size_t ws_size,
                              hipStream_t stream) {
  const float* queries = (const float*)d_in[0];
  const float* keys    = (const float*)d_in[1];
  const float* values  = (const float*)d_in[2];
  const float* Wq  = (const float*)d_in[3];
  const float* bq  = (const float*)d_in[4];
  const float* Wlk = (const float*)d_in[5];
  const float* blk = (const float*)d_in[6];
  const float* Wlv = (const float*)d_in[7];
  const float* blv = (const float*)d_in[8];
  const float* Wkr = (const float*)d_in[9];
  const float* bkr = (const float*)d_in[10];
  const float* Wvr = (const float*)d_in[11];
  const float* bvr = (const float*)d_in[12];
  const float* Wo  = (const float*)d_in[13];
  const float* bo  = (const float*)d_in[14];
  float* out = (float*)d_out;

  char* ws = (char*)d_ws;
  size_t off = 0;
  auto carve = [&](size_t elems) -> __bf16* {
    __bf16* p = (__bf16*)(ws + off);
    off += (elems * sizeof(__bf16) + 255) & ~(size_t)255;
    return p;
  };

  const size_t BS = (size_t)B_ * S_;           // 4096
  __bf16* qbf   = carve(BS * D_);
  __bf16* kbf   = carve(BS * D_);
  __bf16* vbf   = carve(BS * D_);
  __bf16* WqT   = carve((size_t)(DK_ * H_) * D_);   // [N=2048][K=2048]
  __bf16* WlkT  = carve((size_t)(L_ * H_) * D_);    // [1024][2048]
  __bf16* WlvT  = carve((size_t)(L_ * H_) * D_);
  __bf16* WkrT  = carve((size_t)DK_ * L_);          // [128][64]
  __bf16* WvrT  = carve((size_t)DV_ * L_);
  __bf16* WoT   = carve((size_t)D_ * (DV_ * H_));   // [2048][2048]
  __bf16* Qp    = carve(BS * (size_t)(H_ * DK_));   // [B*S, H*DK]
  __bf16* latK  = carve(BS * (size_t)(H_ * L_));    // [B*S, H*L]
  __bf16* latV  = carve(BS * (size_t)(H_ * L_));
  __bf16* Kr    = carve((size_t)B_ * H_ * S_ * DK_); // [B,H,S,DK]
  __bf16* VTr   = carve((size_t)B_ * H_ * DV_ * S_); // [B,H,DV,S]
  __bf16* Ob    = carve(BS * (size_t)(H_ * DV_));    // [B*S, H*DV]

  auto cvt = [&](const float* src, __bf16* dst, size_t n) {
    int n4 = (int)(n / 4);
    cvt_bf16_kernel<<<dim3((n4 + 255) / 256), dim3(256), 0, stream>>>(src, dst, n4);
  };
  cvt(queries, qbf, BS * D_);
  cvt(keys,    kbf, BS * D_);
  cvt(values,  vbf, BS * D_);

  auto trans = [&](const float* W, __bf16* WT, int K, int N) {
    cvt_transpose_kernel<<<dim3(N / 32, K / 32), dim3(256), 0, stream>>>(W, WT, K, N);
  };
  trans(Wq,  WqT,  D_,  DK_ * H_);
  trans(Wlk, WlkT, D_,  L_ * H_);
  trans(Wlv, WlvT, D_,  L_ * H_);
  trans(Wkr, WkrT, L_,  DK_);
  trans(Wvr, WvrT, L_,  DV_);
  trans(Wo,  WoT,  DV_ * H_, D_);

  auto gemm = [&](const __bf16* A, const __bf16* WT, const float* bias,
                  __bf16* Cb, float* Cf, int M, int N, int K,
                  int lda, int ldc,
                  long aSB, long aSH, int hDiv, long cSZ, int tStore, int batch) {
    gemm_bf16_kernel<<<dim3(N / 128, M / 128, batch), dim3(256), 0, stream>>>(
        A, WT, bias, Cb, Cf, M, N, K, lda, ldc, aSB, aSH, hDiv, cSZ, tStore);
  };

  // 1) Q projection: [4096,2048] = qbf @ Wq + bq
  gemm(qbf, WqT, bq, Qp, nullptr, (int)BS, DK_ * H_, D_, D_, DK_ * H_,
       0, 0, 1, 0, 0, 1);
  // 2) latent K / V: [4096,1024]
  gemm(kbf, WlkT, blk, latK, nullptr, (int)BS, L_ * H_, D_, D_, L_ * H_,
       0, 0, 1, 0, 0, 1);
  gemm(vbf, WlvT, blv, latV, nullptr, (int)BS, L_ * H_, D_, D_, L_ * H_,
       0, 0, 1, 0, 0, 1);
  // 3) K reconstruction, batched over z=b*H+h
  gemm(latK, WkrT, bkr, Kr, nullptr, S_, DK_, L_, H_ * L_, DK_,
       (long)S_ * H_ * L_, (long)L_, H_, (long)S_ * DK_, 0, B_ * H_);
  // 4) V reconstruction, stored transposed
  gemm(latV, WvrT, bvr, VTr, nullptr, S_, DV_, L_, H_ * L_, S_,
       (long)S_ * H_ * L_, (long)L_, H_, (long)DV_ * S_, 1, B_ * H_);
  // 5) causal flash attention
  mla_flash_kernel<<<dim3(S_ / 64, B_ * H_), dim3(128), 0, stream>>>(Qp, Kr, VTr, Ob);
  // 6) output projection -> f32 d_out
  gemm(Ob, WoT, bo, nullptr, out, (int)BS, D_, DV_ * H_, DV_ * H_, D_,
       0, 0, 1, 0, 0, 1);
}
